// Attention_52578989638442
// MI455X (gfx1250) — compile-verified, matrix operand order checked by reading, not run on hardware
//
#include <hip/hip_runtime.h>
#include <math.h>

typedef __attribute__((ext_vector_type(16))) __bf16 v16bf;
typedef __attribute__((ext_vector_type(4)))  __bf16 v4bf;
typedef __attribute__((ext_vector_type(8)))  float  v8f;

#define EMB    2048
#define KVD    512
#define NB     2
#define TSEQ   2048
#define BT     (NB * TSEQ)      // 4096 rows
#define NQH    32
#define NKVH   8

__device__ __forceinline__ v8f wmma_bf16(v16bf a, v16bf b, v8f c) {
    return __builtin_amdgcn_wmma_f32_16x16x32_bf16(false, a, false, b,
                                                   (short)0, c, false, false);
}

// ---------------------------------------------------------------------------
// f32 -> bf16 conversion (vectorized x4)
// ---------------------------------------------------------------------------
__global__ void cvt_bf16_kernel(const float* __restrict__ in,
                                __bf16* __restrict__ out, int n4) {
    int i = blockIdx.x * blockDim.x + threadIdx.x;
    if (i < n4) {
        float4 v = ((const float4*)in)[i];
        v4bf o;
        o[0] = (__bf16)v.x; o[1] = (__bf16)v.y;
        o[2] = (__bf16)v.z; o[3] = (__bf16)v.w;
        ((v4bf*)out)[i] = o;
    }
}

// ---------------------------------------------------------------------------
// C[M,N] = A[M,K](bf16) * W[N,K]^T(bf16) + bias, K-contiguous both sides.
// Wave computes 32x64 (2 A-frags x 4 B-frags -> 8 WMMA per 6 loads);
// block = 8 waves (4M x 2N) = 128x128 tile.
// ---------------------------------------------------------------------------
__global__ __launch_bounds__(256)
void gemm_bf16_tn(const __bf16* __restrict__ A, const __bf16* __restrict__ W,
                  const float* __restrict__ bias,
                  float* __restrict__ Cf, __bf16* __restrict__ Cb,
                  int M, int N, int K) {
    const int lane = threadIdx.x & 31;
    const int wv   = threadIdx.x >> 5;
    const int wm   = wv & 3;             // 0..3 (M dir)
    const int wn   = wv >> 2;            // 0..1 (N dir)
    const int lh   = lane >> 4;          // K-half select
    const int ll   = lane & 15;          // m/n within tile
    const int m0   = blockIdx.x * 128 + wm * 32;
    const int n0   = blockIdx.y * 128 + wn * 64;

    const __bf16* arow0 = A + (size_t)(m0 + ll)      * K + lh * 16;
    const __bf16* arow1 = A + (size_t)(m0 + 16 + ll) * K + lh * 16;
    const __bf16* brow0 = W + (size_t)(n0 + ll)      * K + lh * 16;
    const __bf16* brow1 = W + (size_t)(n0 + 16 + ll) * K + lh * 16;
    const __bf16* brow2 = W + (size_t)(n0 + 32 + ll) * K + lh * 16;
    const __bf16* brow3 = W + (size_t)(n0 + 48 + ll) * K + lh * 16;

    v8f acc[2][4] = {};
    for (int kk = 0; kk < K; kk += 32) {
        if (kk + 512 < K) {
            // locality 3 -> WGP-scope prefetch (fills all cache levels)
            __builtin_prefetch(arow0 + kk + 512, 0, 3);
            __builtin_prefetch(arow1 + kk + 512, 0, 3);
            __builtin_prefetch(brow0 + kk + 512, 0, 3);
            __builtin_prefetch(brow2 + kk + 512, 0, 3);
        }
        v16bf a0 = *(const v16bf*)(arow0 + kk);
        v16bf a1 = *(const v16bf*)(arow1 + kk);
        v16bf b0 = *(const v16bf*)(brow0 + kk);
        v16bf b1 = *(const v16bf*)(brow1 + kk);
        v16bf b2 = *(const v16bf*)(brow2 + kk);
        v16bf b3 = *(const v16bf*)(brow3 + kk);
        acc[0][0] = wmma_bf16(a0, b0, acc[0][0]);
        acc[0][1] = wmma_bf16(a0, b1, acc[0][1]);
        acc[0][2] = wmma_bf16(a0, b2, acc[0][2]);
        acc[0][3] = wmma_bf16(a0, b3, acc[0][3]);
        acc[1][0] = wmma_bf16(a1, b0, acc[1][0]);
        acc[1][1] = wmma_bf16(a1, b1, acc[1][1]);
        acc[1][2] = wmma_bf16(a1, b2, acc[1][2]);
        acc[1][3] = wmma_bf16(a1, b3, acc[1][3]);
    }

    // C layout: VGPR r holds rows r (lanes 0-15) / r+8 (lanes 16-31), col = lane%16
    #pragma unroll
    for (int i = 0; i < 2; ++i) {
        #pragma unroll
        for (int j = 0; j < 4; ++j) {
            int n  = n0 + j * 16 + ll;
            float bb = bias ? bias[n] : 0.f;
            #pragma unroll
            for (int r = 0; r < 8; ++r) {
                int m = m0 + i * 16 + r + 8 * lh;
                float v = acc[i][j][r] + bb;
                if (Cf) Cf[(size_t)m * N + n] = v;
                else    Cb[(size_t)m * N + n] = (__bf16)v;
            }
        }
    }
}

// ---------------------------------------------------------------------------
// RoPE: in f32 [B,T,H,64] -> out bf16, rotate-half convention.
// One thread per (row-head, pair i).
// ---------------------------------------------------------------------------
__global__ void rope_bf16_kernel(const float* __restrict__ in,
                                 __bf16* __restrict__ out, int H, int n) {
    int idx = blockIdx.x * blockDim.x + threadIdx.x;
    if (idx >= n) return;
    int i       = idx & 31;
    int rowhead = idx >> 5;                 // (b*T + t)*H + h
    int t       = (rowhead / H) % TSEQ;
    size_t base = (size_t)rowhead * 64;
    float ang = (float)t * __powf(10000.f, -(float)i * (1.0f / 32.0f));
    float s, c;
    __sincosf(ang, &s, &c);
    float lo = in[base + i];
    float hi = in[base + 32 + i];
    out[base + i]      = (__bf16)(lo * c - hi * s);
    out[base + 32 + i] = (__bf16)(hi * c + lo * s);
}

// ---------------------------------------------------------------------------
// Causal GQA flash attention. Block = 4 waves x 16 q-rows = 64 q-rows.
// grid = (B*NQH, T/64). Q/K/V bf16, O accum f32 -> Y bf16 [B,T,NQH,64].
// ---------------------------------------------------------------------------
__global__ __launch_bounds__(128)
void flash_attn_kernel(const __bf16* __restrict__ Q,
                       const __bf16* __restrict__ Kg,
                       const __bf16* __restrict__ Vg,
                       __bf16* __restrict__ Y) {
    __shared__ __bf16 kT[32][80];       // [key][d], padded rows (160B, 32B-mult)
    __shared__ __bf16 vT[64][48];       // [d][key] transposed (96B rows)
    __shared__ __bf16 pT[4][16][48];    // per-wave P staging (C->A relayout)

    const int lane  = threadIdx.x & 31;
    const int wv    = threadIdx.x >> 5;
    const int lh    = lane >> 4;
    const int ll    = lane & 15;
    const int bh    = blockIdx.x;               // b*NQH + h
    const int b     = bh >> 5;
    const int h     = bh & 31;
    const int hkv   = h >> 2;                   // GQA: 4 q-heads per kv-head
    const int qbase = blockIdx.y * 64;
    const int qrow0 = qbase + wv * 16;

    // Q fragments (16 rows x 64 d = two 16x32 A-frags), resident for the kernel
    const __bf16* qptr = Q + ((size_t)(b * TSEQ + qrow0 + ll)) * EMB + h * 64;
    v16bf qa0 = *(const v16bf*)(qptr + lh * 16);
    v16bf qa1 = *(const v16bf*)(qptr + 32 + lh * 16);

    v8f o0{}, o1{}, o2{}, o3{};
    float rmax[8], rsum[8];
    #pragma unroll
    for (int r = 0; r < 8; ++r) { rmax[r] = -__builtin_inff(); rsum[r] = 0.f; }

    const int nkt = (qbase + 64) >> 5;          // 32-key tiles, causal bound
    for (int kt = 0; kt < nkt; ++kt) {
        const int kb0 = kt * 32;
        // cooperative stage of K (row-major) and V (transposed) into LDS
        {
            int row = threadIdx.x >> 2;                 // 0..31
            int ch  = (threadIdx.x & 3) * 16;           // 0/16/32/48
            size_t g = ((size_t)(b * TSEQ + kb0 + row)) * KVD + hkv * 64 + ch;
            v16bf kv = *(const v16bf*)(Kg + g);
            *(v16bf*)&kT[row][ch] = kv;
            v16bf vv = *(const v16bf*)(Vg + g);
            #pragma unroll
            for (int j = 0; j < 16; ++j) vT[ch + j][row] = vv[j];
        }
        __syncthreads();

        // S = Q * K^T  (two 16x16 key halves, K-dim = 64 -> 2 WMMA each)
        v8f s0{}, s1{};
        {
            v16bf kb00 = *(const v16bf*)&kT[ll][lh * 16];
            v16bf kb01 = *(const v16bf*)&kT[ll][32 + lh * 16];
            s0 = wmma_bf16(qa0, kb00, s0);
            s0 = wmma_bf16(qa1, kb01, s0);
            v16bf kb10 = *(const v16bf*)&kT[16 + ll][lh * 16];
            v16bf kb11 = *(const v16bf*)&kT[16 + ll][32 + lh * 16];
            s1 = wmma_bf16(qa0, kb10, s1);
            s1 = wmma_bf16(qa1, kb11, s1);
        }

        // online softmax (row stats replicated across each 16-lane half)
        #pragma unroll
        for (int r = 0; r < 8; ++r) {
            int   qi = qrow0 + r + 8 * lh;
            float v0 = s0[r] * 0.125f;              // 1/sqrt(64)
            float v1 = s1[r] * 0.125f;
            if (kb0 + ll > qi)      v0 = -__builtin_inff();
            if (kb0 + 16 + ll > qi) v1 = -__builtin_inff();
            float mx = fmaxf(v0, v1);
            #pragma unroll
            for (int off = 1; off < 16; off <<= 1)
                mx = fmaxf(mx, __shfl_xor(mx, off, 32));
            float nm   = fmaxf(rmax[r], mx);
            float corr = __expf(rmax[r] - nm);
            float p0   = __expf(v0 - nm);
            float p1   = __expf(v1 - nm);
            float ps   = p0 + p1;
            #pragma unroll
            for (int off = 1; off < 16; off <<= 1)
                ps += __shfl_xor(ps, off, 32);
            rsum[r] = rsum[r] * corr + ps;
            rmax[r] = nm;
            o0[r] *= corr; o1[r] *= corr; o2[r] *= corr; o3[r] *= corr;
            int mrow = r + 8 * lh;
            pT[wv][mrow][ll]      = (__bf16)p0;
            pT[wv][mrow][16 + ll] = (__bf16)p1;
        }

        // O += P * V   (P: 16x32 A-frag via LDS relayout; V^T gives contiguous B-frags)
        {
            v16bf pa  = *(const v16bf*)&pT[wv][ll][lh * 16];
            v16bf vb0 = *(const v16bf*)&vT[ll][lh * 16];
            v16bf vb1 = *(const v16bf*)&vT[16 + ll][lh * 16];
            v16bf vb2 = *(const v16bf*)&vT[32 + ll][lh * 16];
            v16bf vb3 = *(const v16bf*)&vT[48 + ll][lh * 16];
            o0 = wmma_bf16(pa, vb0, o0);
            o1 = wmma_bf16(pa, vb1, o1);
            o2 = wmma_bf16(pa, vb2, o2);
            o3 = wmma_bf16(pa, vb3, o3);
        }
        __syncthreads();
    }

    // normalize and write Y[b, t, h, d]
    #pragma unroll
    for (int r = 0; r < 8; ++r) {
        int    qi  = qrow0 + r + 8 * lh;
        float  inv = 1.f / rsum[r];
        size_t base = ((size_t)(b * TSEQ + qi)) * EMB + h * 64 + ll;
        Y[base]      = (__bf16)(o0[r] * inv);
        Y[base + 16] = (__bf16)(o1[r] * inv);
        Y[base + 32] = (__bf16)(o2[r] * inv);
        Y[base + 48] = (__bf16)(o3[r] * inv);
    }
}

// ---------------------------------------------------------------------------
// Host launcher
// ---------------------------------------------------------------------------
extern "C" void kernel_launch(void* const* d_in, const int* in_sizes, int n_in,
                              void* d_out, int out_size, void* d_ws, size_t ws_size,
                              hipStream_t stream) {
    const float* x  = (const float*)d_in[0];
    const float* Wq = (const float*)d_in[1];
    const float* bq = (const float*)d_in[2];
    const float* Wk = (const float*)d_in[3];
    const float* bk = (const float*)d_in[4];
    const float* Wv = (const float*)d_in[5];
    const float* bv = (const float*)d_in[6];
    const float* Wo = (const float*)d_in[7];
    const float* bo = (const float*)d_in[8];

    char* ws = (char*)d_ws;
    // workspace layout (bytes); ybuf overlays qf, vbuf overlays kf (both dead
    // by the time the overlapping producer runs — stream order serializes).
    __bf16* xb   = (__bf16*)(ws + 0);                         // 16,777,216
    __bf16* Wqb  = (__bf16*)(ws + 16777216);                  //  8,388,608
    __bf16* Wkb  = (__bf16*)(ws + 25165824);                  //  2,097,152
    __bf16* Wvb  = (__bf16*)(ws + 27262976);                  //  2,097,152
    __bf16* Wob  = (__bf16*)(ws + 29360128);                  //  8,388,608
    float*  qf   = (float*) (ws + 37748736);                  // 33,554,432
    float*  kf   = (float*) (ws + 71303168);                  //  8,388,608
    __bf16* qbuf = (__bf16*)(ws + 79691776);                  // 16,777,216
    __bf16* kbuf = (__bf16*)(ws + 96468992);                  //  4,194,304
    __bf16* ybuf = (__bf16*)(ws + 37748736);                  // reuse qf
    __bf16* vbuf = (__bf16*)(ws + 71303168);                  // reuse kf

    // 1) convert inputs to bf16
    cvt_bf16_kernel<<<(BT * EMB / 4 + 255) / 256, 256, 0, stream>>>(x,  xb,  BT * EMB / 4);
    cvt_bf16_kernel<<<(EMB * EMB / 4 + 255) / 256, 256, 0, stream>>>(Wq, Wqb, EMB * EMB / 4);
    cvt_bf16_kernel<<<(KVD * EMB / 4 + 255) / 256, 256, 0, stream>>>(Wk, Wkb, KVD * EMB / 4);
    cvt_bf16_kernel<<<(KVD * EMB / 4 + 255) / 256, 256, 0, stream>>>(Wv, Wvb, KVD * EMB / 4);
    cvt_bf16_kernel<<<(EMB * EMB / 4 + 255) / 256, 256, 0, stream>>>(Wo, Wob, EMB * EMB / 4);

    // 2) Q/K projections -> f32 (RoPE input)
    dim3 gq(BT / 128, EMB / 128);
    dim3 gk(BT / 128, KVD / 128);
    gemm_bf16_tn<<<gq, 256, 0, stream>>>(xb, Wqb, bq, qf, nullptr, BT, EMB, EMB);
    gemm_bf16_tn<<<gk, 256, 0, stream>>>(xb, Wkb, bk, kf, nullptr, BT, KVD, EMB);

    // 3) RoPE -> bf16 Q/K   (frees qf/kf regions for reuse)
    rope_bf16_kernel<<<(BT * NQH * 32) / 256, 256, 0, stream>>>(qf, qbuf, NQH, BT * NQH * 32);
    rope_bf16_kernel<<<(BT * NKVH * 32) / 256, 256, 0, stream>>>(kf, kbuf, NKVH, BT * NKVH * 32);

    // 4) V projection straight to bf16 (overlays kf)
    gemm_bf16_tn<<<gk, 256, 0, stream>>>(xb, Wvb, bv, nullptr, vbuf, BT, KVD, EMB);

    // 5) flash attention (overlays qf with ybuf)
    dim3 ga(NB * NQH, TSEQ / 64);
    flash_attn_kernel<<<ga, 128, 0, stream>>>(qbuf, kbuf, vbuf, ybuf);

    // 6) output projection -> f32 d_out
    gemm_bf16_tn<<<gq, 256, 0, stream>>>(ybuf, Wob, bo, (float*)d_out, nullptr, BT, EMB, EMB);
}
